// VimBlock_38465727103455
// MI455X (gfx1250) — matrix operand order repaired and checked by reading, not compiled, not run on hardware
//
#include <hip/hip_runtime.h>

// ---------------- problem constants ----------------
#define DIMC   192
#define DINNER 384
#define DTRANK 12
#define DSTATE 16
#define LSEQ   4096
#define NB     4
#define KC     4     // conv kernel
#define NXP    44    // dt_rank + 2*d_state
#define NXPP   48    // padded to 3 wmma tiles

typedef __attribute__((ext_vector_type(16))) _Float16 v16h;
typedef __attribute__((ext_vector_type(8)))  _Float16 h8;
typedef __attribute__((ext_vector_type(8)))  float    v8f;

// ---------------- WMMA fragment loaders (f16, 16x16x32) ----------------
// A (16x32, row-major source, lda in halves):
//   lane l: m = l&15, hi = (l>>4)*8 ; elems 0..7 = K hi..hi+7 ; 8..15 = K 16+hi..
__device__ __forceinline__ v16h load_a_frag(const _Float16* __restrict__ A, int lda,
                                            int m0, int k0, int lane) {
  const _Float16* p = A + (size_t)(m0 + (lane & 15)) * lda + k0 + ((lane >> 4) << 3);
  h8 x0 = *(const h8*)(p);
  h8 x1 = *(const h8*)(p + 16);
  return __builtin_shufflevector(x0, x1, 0,1,2,3,4,5,6,7,8,9,10,11,12,13,14,15);
}
// B (32x16) from weights stored (N,K) row-major:
//   lane l: n = l&15, hi = (l>>4)*8 ; elems 0..7 = K hi..hi+7 ; 8..15 = K 16+hi..
__device__ __forceinline__ v16h load_b_frag(const _Float16* __restrict__ W, int ldb,
                                            int n0, int k0, int lane) {
  const _Float16* p = W + (size_t)(n0 + (lane & 15)) * ldb + k0 + ((lane >> 4) << 3);
  h8 x0 = *(const h8*)(p);
  h8 x1 = *(const h8*)(p + 16);
  return __builtin_shufflevector(x0, x1, 0,1,2,3,4,5,6,7,8,9,10,11,12,13,14,15);
}

#define WMMA_F16(a, b, c) \
  __builtin_amdgcn_wmma_f32_16x16x32_f16(false, (a), false, (b), (short)0, (c), false, false)

// wave-level MMA core: acc[MT][NT] += A[m0:+MT*16, :K] * W[n0:+NT*16, :K]^T
template<int MT, int NT>
__device__ __forceinline__ void wave_mma(const _Float16* __restrict__ A, int lda,
                                         const _Float16* __restrict__ W, int ldb,
                                         int K, int m0, int n0, int lane,
                                         v8f acc[MT][NT]) {
  for (int k0 = 0; k0 < K; k0 += 32) {
    v16h af[MT];
#pragma unroll
    for (int mt = 0; mt < MT; ++mt) af[mt] = load_a_frag(A, lda, m0 + mt * 16, k0, lane);
#pragma unroll
    for (int nt = 0; nt < NT; ++nt) {
      v16h bf = load_b_frag(W, ldb, n0 + nt * 16, k0, lane);
#pragma unroll
      for (int mt = 0; mt < MT; ++mt) acc[mt][nt] = WMMA_F16(af[mt], bf, acc[mt][nt]);
    }
  }
}

// ---------------- stage 0: transpose x (B,C,L) -> xT (B,L,C) as f16 ----------------
__global__ void transpose_kernel(const float* __restrict__ x, _Float16* __restrict__ xT) {
  __shared__ float tile[32][33];
  int cb = blockIdx.x << 5, lb = blockIdx.y << 5, b = blockIdx.z;
  int tx = threadIdx.x & 31, ty = threadIdx.x >> 5;
#pragma unroll
  for (int i = ty; i < 32; i += 8)
    tile[i][tx] = x[((size_t)b * DIMC + cb + i) * LSEQ + lb + tx];
  __syncthreads();
#pragma unroll
  for (int i = ty; i < 32; i += 8)
    xT[((size_t)b * LSEQ + lb + i) * DIMC + cb + tx] = (_Float16)tile[tx][i];
}

// ---------------- helpers: f32->f16 weight conversion / padding ----------------
__global__ void cvt_f16_kernel(const float* __restrict__ in, _Float16* __restrict__ out, int n) {
  int i = blockIdx.x * blockDim.x + threadIdx.x;
  if (i < n) out[i] = (_Float16)in[i];
}
__global__ void pad_wx_kernel(const float* __restrict__ Wx, _Float16* __restrict__ Wxp) {
  int i = blockIdx.x * blockDim.x + threadIdx.x;   // over 48*384
  Wxp[i] = (i < NXP * DINNER) ? (_Float16)Wx[i] : (_Float16)0.f;
}

// ---------------- stage 1: in_proj GEMM  xz = xT @ W_in^T  (M=16384,N=768,K=192) ----
// wave tile 64x32 (MT=4, NT=2): 8 f32 accum frags = 64 VGPRs -> no spills
__global__ void __launch_bounds__(256, 1)
gemm_in_kernel(const _Float16* __restrict__ A, const _Float16* __restrict__ W,
               float* __restrict__ xin, float* __restrict__ zg) {
  const int lane = threadIdx.x & 31;
  const int wg = (blockIdx.x * blockDim.x + threadIdx.x) >> 5;
  const int tm = wg / 24, tn = wg % 24;          // 256 x 24 wave tiles of 64x32
  const int m0 = tm * 64, n0 = tn * 32;
  v8f acc[4][2];
#pragma unroll
  for (int mt = 0; mt < 4; ++mt)
#pragma unroll
    for (int nt = 0; nt < 2; ++nt)
#pragma unroll
      for (int r = 0; r < 8; ++r) acc[mt][nt][r] = 0.f;

  wave_mma<4, 2>(A, DIMC, W, DIMC, DIMC, m0, n0, lane, acc);

  const int nl = lane & 15, mh = (lane >> 4) * 8;
#pragma unroll
  for (int mt = 0; mt < 4; ++mt)
#pragma unroll
    for (int nt = 0; nt < 2; ++nt)
#pragma unroll
      for (int r = 0; r < 8; ++r) {
        int m = m0 + mt * 16 + mh + r;
        int n = n0 + nt * 16 + nl;
        float v = acc[mt][nt][r];
        if (n < DINNER) xin[(size_t)m * DINNER + n] = v;
        else            zg[(size_t)m * DINNER + (n - DINNER)] = v;
      }
}

// ---------------- stage 2: depthwise causal conv (k=4) + SiLU ----------------
__global__ void conv_silu_kernel(const float* __restrict__ xin, const float* __restrict__ w,
                                 const float* __restrict__ bias, float* __restrict__ xcf,
                                 _Float16* __restrict__ xch) {
  size_t idx = (size_t)blockIdx.x * blockDim.x + threadIdx.x;  // over M*DINNER
  int d = (int)(idx % DINNER);
  size_t m = idx / DINNER;
  int l = (int)(m & (LSEQ - 1));
  float acc = bias[d];
#pragma unroll
  for (int j = 0; j < KC; ++j) {
    int lj = l - (KC - 1) + j;
    if (lj >= 0) acc += w[d * KC + j] * xin[(m - (size_t)(KC - 1 - j)) * DINNER + d];
  }
  float s = acc / (1.f + __expf(-acc));   // SiLU
  xcf[idx] = s;
  xch[idx] = (_Float16)s;
}

// ---------------- stage 3: x_proj GEMM  x_dbl = x_conv @ W_x^T (N padded 44->48) ----
// wave tile 32x48 (MT=2, NT=3)
__global__ void __launch_bounds__(256, 1)
gemm_xproj_kernel(const _Float16* __restrict__ A, const _Float16* __restrict__ W,
                  float* __restrict__ dtr, float* __restrict__ Bs, float* __restrict__ Cs) {
  const int lane = threadIdx.x & 31;
  const int wg = (blockIdx.x * blockDim.x + threadIdx.x) >> 5;  // 512 waves, 32-row slabs
  const int m0 = wg * 32;
  v8f acc[2][3];
#pragma unroll
  for (int mt = 0; mt < 2; ++mt)
#pragma unroll
    for (int nt = 0; nt < 3; ++nt)
#pragma unroll
      for (int r = 0; r < 8; ++r) acc[mt][nt][r] = 0.f;

  wave_mma<2, 3>(A, DINNER, W, DINNER, DINNER, m0, 0, lane, acc);

  const int nl = lane & 15, mh = (lane >> 4) * 8;
#pragma unroll
  for (int mt = 0; mt < 2; ++mt)
#pragma unroll
    for (int nt = 0; nt < 3; ++nt)
#pragma unroll
      for (int r = 0; r < 8; ++r) {
        int m = m0 + mt * 16 + mh + r;
        int n = nt * 16 + nl;
        float v = acc[mt][nt][r];
        if (n < DTRANK)                dtr[(size_t)m * DTRANK + n] = v;
        else if (n < DTRANK + DSTATE)  Bs[(size_t)m * DSTATE + (n - DTRANK)] = v;
        else if (n < NXP)              Cs[(size_t)m * DSTATE + (n - DTRANK - DSTATE)] = v;
      }
}

// ---------------- stage 4: dt_proj + softplus (double bias, faithful to ref) -------
__global__ void dtproj_kernel(const float* __restrict__ dtr, const float* __restrict__ Wdt,
                              const float* __restrict__ bdt, float* __restrict__ delta) {
  size_t idx = (size_t)blockIdx.x * blockDim.x + threadIdx.x;
  int d = (int)(idx % DINNER);
  size_t m = idx / DINNER;
  float acc = 2.f * bdt[d];
#pragma unroll
  for (int r = 0; r < DTRANK; ++r) acc += dtr[m * DTRANK + r] * Wdt[d * DTRANK + r];
  delta[idx] = (acc > 20.f) ? acc : log1pf(__expf(acc));
}

// ---------------- stage 5: selective scan, fused D-skip + SiLU gate ----------------
// one lane per independent recurrence (b, d, n); 16-lane reduction over n
__global__ void scan_kernel(const float* __restrict__ delta, const float* __restrict__ u,
                            const float* __restrict__ Bs, const float* __restrict__ Cs,
                            const float* __restrict__ zg, const float* __restrict__ A_log,
                            const float* __restrict__ Dv, _Float16* __restrict__ y) {
  int b = blockIdx.x / (DINNER / 16);
  int dblk = blockIdx.x % (DINNER / 16);
  int n = threadIdx.x & 15;
  int d = dblk * 16 + (threadIdx.x >> 4);
  float Adn = -__expf(A_log[d * DSTATE + n]);
  float Dd = Dv[d];
  float h = 0.f;
  size_t mb = (size_t)b * LSEQ;
  for (int t = 0; t < LSEQ; ++t) {
    size_t m = mb + t;
    if (t + 16 < LSEQ) {                                    // stream ahead (global_prefetch_b8)
      __builtin_prefetch(&delta[(m + 16) * DINNER + d], 0, 1);
      __builtin_prefetch(&u[(m + 16) * DINNER + d], 0, 1);
    }
    float dt = delta[m * DINNER + d];
    float ut = u[m * DINNER + d];
    float Bn = Bs[m * DSTATE + n];
    float Cn = Cs[m * DSTATE + n];
    h = __expf(dt * Adn) * h + (dt * ut) * Bn;              // per-state recurrence
    float p = h * Cn;                                       // reduce over 16 states
    p += __shfl_xor(p, 1, 16);
    p += __shfl_xor(p, 2, 16);
    p += __shfl_xor(p, 4, 16);
    p += __shfl_xor(p, 8, 16);
    if (n == 0) {
      float zv = zg[m * DINNER + d];
      float sil = zv / (1.f + __expf(-zv));
      y[m * DINNER + d] = (_Float16)((p + Dd * ut) * sil);
    }
  }
}

// ---------------- stage 6: out_proj GEMM -> out (B,C,L) ----------------
// wave tile 64x32 (MT=4, NT=2)
__global__ void __launch_bounds__(256, 1)
gemm_out_kernel(const _Float16* __restrict__ A, const _Float16* __restrict__ W,
                float* __restrict__ out) {
  const int lane = threadIdx.x & 31;
  const int wg = (blockIdx.x * blockDim.x + threadIdx.x) >> 5;
  const int tm = wg / 6, tn = wg % 6;            // 256 x 6 wave tiles of 64x32
  const int m0 = tm * 64, n0 = tn * 32;
  v8f acc[4][2];
#pragma unroll
  for (int mt = 0; mt < 4; ++mt)
#pragma unroll
    for (int nt = 0; nt < 2; ++nt)
#pragma unroll
      for (int r = 0; r < 8; ++r) acc[mt][nt][r] = 0.f;

  wave_mma<4, 2>(A, DINNER, W, DINNER, DINNER, m0, n0, lane, acc);

  const int nl = lane & 15, mh = (lane >> 4) * 8;
#pragma unroll
  for (int mt = 0; mt < 4; ++mt)
#pragma unroll
    for (int nt = 0; nt < 2; ++nt)
#pragma unroll
      for (int r = 0; r < 8; ++r) {
        int m = m0 + mt * 16 + mh + r;
        int n = n0 + nt * 16 + nl;                 // n = output channel c
        int bidx = m >> 12, l = m & (LSEQ - 1);
        out[((size_t)bidx * DIMC + n) * LSEQ + l] = acc[mt][nt][r];
      }
}

// ---------------- launcher ----------------
extern "C" void kernel_launch(void* const* d_in, const int* in_sizes, int n_in,
                              void* d_out, int out_size, void* d_ws, size_t ws_size,
                              hipStream_t stream) {
  (void)in_sizes; (void)n_in; (void)out_size; (void)ws_size;
  const float* x     = (const float*)d_in[0];
  const float* W_in  = (const float*)d_in[1];
  const float* convw = (const float*)d_in[2];
  const float* convb = (const float*)d_in[3];
  const float* W_x   = (const float*)d_in[4];
  const float* W_dt  = (const float*)d_in[5];
  const float* b_dt  = (const float*)d_in[6];
  const float* A_log = (const float*)d_in[7];
  const float* Dv    = (const float*)d_in[8];
  const float* W_out = (const float*)d_in[9];
  float* out = (float*)d_out;

  const size_t M = (size_t)NB * LSEQ;            // 16384 rows

  char* ws = (char*)d_ws;
  size_t off = 0;
  auto allocB = [&](size_t bytes) -> char* {
    char* p = ws + off;
    off += (bytes + 15) & ~(size_t)15;
    return p;
  };
  _Float16* xT    = (_Float16*)allocB(M * DIMC * 2);
  _Float16* Winh  = (_Float16*)allocB((size_t)2 * DINNER * DIMC * 2);
  _Float16* Wxph  = (_Float16*)allocB((size_t)NXPP * DINNER * 2);
  _Float16* Wouth = (_Float16*)allocB((size_t)DIMC * DINNER * 2);
  _Float16* xch   = (_Float16*)allocB(M * DINNER * 2);
  _Float16* ybuf  = (_Float16*)allocB(M * DINNER * 2);
  float* xin   = (float*)allocB(M * DINNER * 4);
  float* zgb   = (float*)allocB(M * DINNER * 4);
  float* xcf   = (float*)allocB(M * DINNER * 4);
  float* dtr   = (float*)allocB(M * DTRANK * 4);
  float* Bsb   = (float*)allocB(M * DSTATE * 4);
  float* Csb   = (float*)allocB(M * DSTATE * 4);
  float* delta = (float*)allocB(M * DINNER * 4);

  // stage 0: layout + precision prep
  transpose_kernel<<<dim3(DIMC / 32, LSEQ / 32, NB), 256, 0, stream>>>(x, xT);
  cvt_f16_kernel<<<(2 * DINNER * DIMC + 255) / 256, 256, 0, stream>>>(W_in, Winh, 2 * DINNER * DIMC);
  pad_wx_kernel<<<(NXPP * DINNER) / 256, 256, 0, stream>>>(W_x, Wxph);
  cvt_f16_kernel<<<(DIMC * DINNER + 255) / 256, 256, 0, stream>>>(W_out, Wouth, DIMC * DINNER);

  // stage 1: in_proj (WMMA): 256*24 wave tiles, 8 waves/block
  gemm_in_kernel<<<(256 * 24) / 8, 256, 0, stream>>>(xT, Winh, xin, zgb);
  // stage 2: depthwise conv + SiLU
  conv_silu_kernel<<<(int)((M * DINNER) / 256), 256, 0, stream>>>(xin, convw, convb, xcf, xch);
  // stage 3: x_proj (WMMA): 512 wave tiles
  gemm_xproj_kernel<<<512 / 8, 256, 0, stream>>>(xch, Wxph, dtr, Bsb, Csb);
  // stage 4: dt_proj + softplus
  dtproj_kernel<<<(int)((M * DINNER) / 256), 256, 0, stream>>>(dtr, W_dt, b_dt, delta);
  // stage 5: selective scan (fused skip + gate)
  scan_kernel<<<NB * (DINNER / 16), 256, 0, stream>>>(delta, xcf, Bsb, Csb, zgb, A_log, Dv, ybuf);
  // stage 6: out_proj (WMMA): 256*6 wave tiles
  gemm_out_kernel<<<(256 * 6) / 8, 256, 0, stream>>>(ybuf, Wouth, out);
}
